// MambaClassifier_36627481100328
// MI455X (gfx1250) — matrix operand order repaired and checked by reading, not compiled, not run on hardware
//
#include <hip/hip_runtime.h>
#include <hip/hip_bf16.h>
#include <math.h>

// ---------------------------------------------------------------------------
// Mamba classifier: B=8, L=4096, DIN=64, D=256, NL=2, DI=512, DS=16, DC=4,
// DTR=16.  Dense projections -> v_wmma_f32_16x16x32_f16 with compile-time
// (NT,K,LDA,N,LDC) so the K loop fully unrolls with zero guards.  Blocks of
// 8 waves share one 16xK A-tile staged in LDS via the CDNA5 async LDS-copy
// path (ASYNCcnt) when the builtin exists, else b128 load+ds_store.
// Selective scan: one persistent block per batch element, state in registers.
// ---------------------------------------------------------------------------

namespace {
constexpr int BB   = 8;
constexpr int LL   = 4096;
constexpr int DIN  = 64;
constexpr int DD   = 256;
constexpr int NLAY = 2;
constexpr int DI   = 512;
constexpr int DS   = 16;
constexpr int DC   = 4;
constexpr int DTR  = 16;
constexpr int MR   = BB * LL;        // 32768 GEMM rows
constexpr int XZS  = 2 * DI;         // 1024: row stride of the xz buffer
constexpr int DBCS = DTR + 2 * DS;   // 48: row stride of dt/B/C buffer
}

typedef __attribute__((ext_vector_type(16))) _Float16 v16h;
typedef __attribute__((ext_vector_type(8)))  float    v8f;

#if defined(__has_builtin)
#if __has_builtin(__builtin_amdgcn_global_load_async_to_lds_b128) && \
    __has_builtin(__builtin_amdgcn_s_wait_asynccnt)
#define USE_ASYNC_LDS 1
#endif
#endif
#ifndef USE_ASYNC_LDS
#define USE_ASYNC_LDS 0
#endif

#if USE_ASYNC_LDS
// Exact parameter types per hipcc diagnostic: AS(1)/AS(3) pointers to a
// 16-byte int vector.
typedef int v4i_async __attribute__((vector_size(16)));
typedef __attribute__((address_space(1))) v4i_async* g4ptr_t;
typedef __attribute__((address_space(3))) v4i_async* l4ptr_t;
typedef __attribute__((address_space(1))) void* g0ptr_t;
typedef __attribute__((address_space(3))) void* l0ptr_t;
#endif

__device__ __forceinline__ float softplus_f(float x) {
  return (x > 20.f) ? x : log1pf(__expf(x));
}
__device__ __forceinline__ float silu_f(float x) {
  return x * (1.f / (1.f + __expf(-x)));
}

__device__ __forceinline__ void cvt8(v16h& v, int base, const float4& a,
                                     const float4& b) {
  v[base + 0] = (_Float16)a.x; v[base + 1] = (_Float16)a.y;
  v[base + 2] = (_Float16)a.z; v[base + 3] = (_Float16)a.w;
  v[base + 4] = (_Float16)b.x; v[base + 5] = (_Float16)b.y;
  v[base + 6] = (_Float16)b.z; v[base + 7] = (_Float16)b.w;
}

// ---------------------------------------------------------------------------
// WMMA GEMM:  C[MR,N] = act( A[MR,K] @ W[N,K]^T + bias )
// SHA=true: 8 waves/block share one m-tile; 16xK A tile staged in LDS
// (async-to-LDS when available), fragments rebuilt via ds_load_b128.
// A fragment (16-bit A 16x32, k = 16h + s as in B): lane L(<16) row M=L, elems 0..15 ->
// K=k0..k0+15; lanes 16..31 take K=k0+16..k0+31.
// B (32x16): lane -> column N=lane%16, elems i -> K=k0+(lane/16)*16+i.
// ---------------------------------------------------------------------------
template <int NT, int K, int LDA, int N, int LDC, bool SHA, bool SOFTPLUS>
__global__ __launch_bounds__(256) void wmma_gemm_kernel(
    const float* __restrict__ A, const float* __restrict__ W,
    const float* __restrict__ bias, float* __restrict__ C) {
  constexpr int NTILES = N / (16 * NT);
  static_assert(N % (16 * NT) == 0, "N tile");
  static_assert(!SHA || (NTILES % 8 == 0), "shared-A needs 8 n-groups/block");
  static_assert(K % 32 == 0 || K == 16, "K shape");
  constexpr int SP = K + 4;  // padded LDS row stride (floats)

  const int lane = threadIdx.x & 31;
  const int half = lane >> 4;  // 0: lanes 0-15, 1: lanes 16-31
  const int lp   = lane & 15;
  const int wid  = threadIdx.x >> 5;

  int mt, n0base;
  if constexpr (SHA) {
    constexpr int NPB = NTILES / 8;
    mt = blockIdx.x / NPB;
    n0base = ((blockIdx.x % NPB) * 8 + wid) * (16 * NT);
  } else {
    const int wave = blockIdx.x * 8 + wid;
    mt = wave / NTILES;
    n0base = (wave % NTILES) * (16 * NT);
  }
  const int m0 = mt * 16;

  __shared__ float sA[SHA ? 16 * SP : 4];

  if constexpr (SHA) {
    // cooperative stage of the 16 x K A-tile (16B chunks per lane)
    constexpr int CH = (16 * K) / 4;
    for (int c = threadIdx.x; c < CH; c += 256) {
      const int r  = c / (K / 4);
      const int k4 = (c % (K / 4)) * 4;
      const float* g = A + (size_t)(m0 + r) * LDA + k4;
      float* l = &sA[r * SP + k4];
#if USE_ASYNC_LDS
      __builtin_amdgcn_global_load_async_to_lds_b128(
          (g4ptr_t)(g0ptr_t)(void*)const_cast<float*>(g),
          (l4ptr_t)(l0ptr_t)(void*)l, 0, 0);
#else
      *reinterpret_cast<float4*>(l) = *reinterpret_cast<const float4*>(g);
#endif
    }
#if USE_ASYNC_LDS
    __builtin_amdgcn_s_wait_asynccnt(0);
#endif
    __syncthreads();
  }

  v8f acc[NT];
  for (int j = 0; j < NT; ++j)
    for (int r = 0; r < 8; ++r) acc[j][r] = 0.f;

  const float* __restrict__ arow =
      SHA ? nullptr : (A + (size_t)(m0 + lp) * LDA);

  if constexpr (K % 32 == 0) {
#pragma unroll
    for (int k0 = 0; k0 < K; k0 += 32) {
      // ---- A fragment (16x32 f16) ----
      v16h af;
      {
        const int s0 = k0 + half * 16;
        const int s1 = k0 + half * 16 + 8;
        const float* base = SHA ? &sA[lp * SP] : arow;
        const float4* p0 = reinterpret_cast<const float4*>(base + s0);
        const float4* p1 = reinterpret_cast<const float4*>(base + s1);
        cvt8(af, 0, p0[0], p0[1]);
        cvt8(af, 8, p1[0], p1[1]);
      }
      // ---- NT B fragments, WMMA accumulate ----
      const int bs = k0 + half * 16;
#pragma unroll
      for (int j = 0; j < NT; ++j) {
        const float* wrow = W + (size_t)(n0base + j * 16 + lp) * K + bs;
        if (j == 0 && k0 + 32 < K) __builtin_prefetch(wrow + 32, 0, 3);
        const float4* p = reinterpret_cast<const float4*>(wrow);
        float4 q0 = p[0], q1 = p[1], q2 = p[2], q3 = p[3];
        v16h bfv;
        cvt8(bfv, 0, q0, q1);
        cvt8(bfv, 8, q2, q3);
        acc[j] = __builtin_amdgcn_wmma_f32_16x16x32_f16(
            false, af, false, bfv, (short)0, acc[j], false, false);
      }
    }
  } else {  // K == 16: single WMMA, upper K-half zero
    v16h af;
    {
      const float* base = SHA ? &sA[lp * SP] : arow;
      const float4* p0 = reinterpret_cast<const float4*>(base);
      cvt8(af, 0, p0[0], p0[1]);
      cvt8(af, 8, p0[2], p0[3]);
    }
#pragma unroll
    for (int j = 0; j < NT; ++j) {
      v16h bfv;
      if (half == 0) {  // lanes 16..31 would read K=16..31 -> zero
        const float4* p = reinterpret_cast<const float4*>(
            W + (size_t)(n0base + j * 16 + lp) * K);
        float4 q0 = p[0], q1 = p[1], q2 = p[2], q3 = p[3];
        cvt8(bfv, 0, q0, q1);
        cvt8(bfv, 8, q2, q3);
      } else {
        for (int i = 0; i < 16; ++i) bfv[i] = (_Float16)0.f;
      }
      acc[j] = __builtin_amdgcn_wmma_f32_16x16x32_f16(
          false, af, false, bfv, (short)0, acc[j], false, false);
    }
  }

  // ---- store (C/D layout: VGPR r -> M = m0 + half*8 + r, N = n0 + lp) ----
#pragma unroll
  for (int j = 0; j < NT; ++j) {
    const int n = n0base + j * 16 + lp;
    const float bv = bias ? bias[n] : 0.f;
#pragma unroll
    for (int r = 0; r < 8; ++r) {
      const int m = m0 + half * 8 + r;
      float v = acc[j][r] + bv;
      if (SOFTPLUS) v = softplus_f(v);
      C[(size_t)m * LDC + n] = v;
    }
  }
}

// ---------------------------------------------------------------------------
// Depthwise causal conv (DC=4) + SiLU, reading the u-half of the xz buffer.
// ---------------------------------------------------------------------------
__global__ __launch_bounds__(256) void conv_silu_kernel(
    const float* __restrict__ xz, const float* __restrict__ cw,
    const float* __restrict__ cb, float* __restrict__ u2, int total) {
  int idx = blockIdx.x * blockDim.x + threadIdx.x;
  const int stride = gridDim.x * blockDim.x;
  for (; idx < total; idx += stride) {
    const int d = idx % DI;
    const int t = (idx / DI) % LL;
    const int b = idx / (DI * LL);
    const float w0 = cw[d * DC + 0], w1 = cw[d * DC + 1];
    const float w2 = cw[d * DC + 2], w3 = cw[d * DC + 3];
    const size_t base = ((size_t)b * LL) * XZS + d;
    float s = cb[d];
    if (t >= 3) s += xz[base + (size_t)(t - 3) * XZS] * w0;
    if (t >= 2) s += xz[base + (size_t)(t - 2) * XZS] * w1;
    if (t >= 1) s += xz[base + (size_t)(t - 1) * XZS] * w2;
    s += xz[base + (size_t)t * XZS] * w3;
    u2[(size_t)(b * LL + t) * DI + d] = silu_f(s);
  }
}

// ---------------------------------------------------------------------------
// Selective scan: one block per batch element, thread d owns the DS=16 state
// slots in registers; Bm/Cm staged through double-buffered LDS (one barrier
// per timestep).  Fuses D skip-term and silu(z) gate; writes over z in xz.
// ---------------------------------------------------------------------------
__global__ __launch_bounds__(512) void scan_kernel(
    const float* __restrict__ u2, const float* __restrict__ dtb,
    const float* __restrict__ dbc, float* __restrict__ xz,
    const float* __restrict__ A_log, const float* __restrict__ Dp) {
  __shared__ float sB[2][DS];
  __shared__ float sC[2][DS];
  const int b = blockIdx.x;
  const int d = threadIdx.x;  // 0..DI-1

  float Av[DS];
  for (int s = 0; s < DS; ++s) Av[s] = -__expf(A_log[d * DS + s]);
  const float Dv = Dp[d];
  float hst[DS];
  for (int s = 0; s < DS; ++s) hst[s] = 0.f;

  for (int t = 0; t < LL; ++t) {
    const int buf = t & 1;
    const size_t row = (size_t)b * LL + t;
    if (threadIdx.x < 2 * DS) {
      const int s = threadIdx.x;
      const float v = dbc[row * DBCS + DTR + s];
      if (s < DS) sB[buf][s] = v;
      else        sC[buf][s - DS] = v;
    }
    __syncthreads();
    const float dt = dtb[row * DI + d];
    const float uu = u2[row * DI + d];
    const float du = dt * uu;
    float y = 0.f;
#pragma unroll
    for (int s = 0; s < DS; ++s) {
      const float dA = __expf(dt * Av[s]);
      hst[s] = hst[s] * dA + du * sB[buf][s];
      y += hst[s] * sC[buf][s];
    }
    const size_t zi = row * XZS + DI + d;
    const float zv = xz[zi];
    xz[zi] = (y + uu * Dv) * silu_f(zv);
  }
}

// ---------------------------------------------------------------------------
// LayerNorm over D=256 (one block per row).
// ---------------------------------------------------------------------------
__global__ __launch_bounds__(256) void layernorm_kernel(
    const float* __restrict__ x, const float* __restrict__ g,
    const float* __restrict__ bta, float* __restrict__ out) {
  __shared__ float red[256];
  const int tid = threadIdx.x;
  const size_t row = blockIdx.x;
  const float v = x[row * DD + tid];
  red[tid] = v;
  __syncthreads();
  for (int off = 128; off > 0; off >>= 1) {
    if (tid < off) red[tid] += red[tid + off];
    __syncthreads();
  }
  const float mean = red[0] * (1.f / DD);
  __syncthreads();
  const float diff = v - mean;
  red[tid] = diff * diff;
  __syncthreads();
  for (int off = 128; off > 0; off >>= 1) {
    if (tid < off) red[tid] += red[tid + off];
    __syncthreads();
  }
  const float var = red[0] * (1.f / DD);
  out[row * DD + tid] = diff * rsqrtf(var + 1e-5f) * g[tid] + bta[tid];
}

// ---------------------------------------------------------------------------
// Attention pooling + linear head: one block per batch element.
// ---------------------------------------------------------------------------
__global__ __launch_bounds__(256) void pool_head_kernel(
    const float* __restrict__ h, const float* __restrict__ wa,
    const float* __restrict__ ba, const float* __restrict__ Wf,
    const float* __restrict__ bf, float* __restrict__ out) {
  __shared__ float logit[LL];  // 16 KB
  __shared__ float red[256];
  const int b = blockIdx.x, tid = threadIdx.x;
  const float* __restrict__ hb = h + (size_t)b * LL * DD;

  for (int t = tid; t < LL; t += 256) {
    const float* __restrict__ row = hb + (size_t)t * DD;
    float s = 0.f;
    for (int d = 0; d < DD; ++d) s += row[d] * wa[d];
    logit[t] = s + ba[0];
  }
  __syncthreads();

  float m = -1e30f;
  for (int t = tid; t < LL; t += 256) m = fmaxf(m, logit[t]);
  red[tid] = m;
  __syncthreads();
  for (int off = 128; off > 0; off >>= 1) {
    if (tid < off) red[tid] = fmaxf(red[tid], red[tid + off]);
    __syncthreads();
  }
  m = red[0];
  __syncthreads();

  float se = 0.f;
  for (int t = tid; t < LL; t += 256) {
    const float e = __expf(logit[t] - m);
    logit[t] = e;
    se += e;
  }
  red[tid] = se;
  __syncthreads();
  for (int off = 128; off > 0; off >>= 1) {
    if (tid < off) red[tid] += red[tid + off];
    __syncthreads();
  }
  const float inv = 1.f / red[0];
  __syncthreads();

  // pooled[d] for d = tid (DD == 256 == blockDim)
  float p = 0.f;
  for (int t = 0; t < LL; ++t) p += logit[t] * hb[(size_t)t * DD + tid];
  p *= inv;
  red[tid] = p * Wf[tid];
  __syncthreads();
  for (int off = 128; off > 0; off >>= 1) {
    if (tid < off) red[tid] += red[tid + off];
    __syncthreads();
  }
  if (tid == 0) out[b] = red[0] + bf[0];
}

// ---------------------------------------------------------------------------
extern "C" void kernel_launch(void* const* d_in, const int* in_sizes, int n_in,
                              void* d_out, int out_size, void* d_ws,
                              size_t ws_size, hipStream_t stream) {
  (void)in_sizes; (void)n_in; (void)out_size; (void)ws_size;
  const float* x    = (const float*)d_in[0];
  const float* Wp   = (const float*)d_in[1];
  const float* bp   = (const float*)d_in[2];
  const float* g0   = (const float*)d_in[3];
  const float* b0   = (const float*)d_in[4];
  const float* Wi   = (const float*)d_in[5];
  const float* cw   = (const float*)d_in[6];
  const float* cb   = (const float*)d_in[7];
  const float* Wx   = (const float*)d_in[8];
  const float* Wdt  = (const float*)d_in[9];
  const float* bdt  = (const float*)d_in[10];
  const float* Alog = (const float*)d_in[11];
  const float* Dp   = (const float*)d_in[12];
  const float* Wo   = (const float*)d_in[13];
  const float* lng  = (const float*)d_in[14];
  const float* lnb  = (const float*)d_in[15];
  const float* wa   = (const float*)d_in[16];
  const float* ba   = (const float*)d_in[17];
  const float* Wf   = (const float*)d_in[18];
  const float* bf   = (const float*)d_in[19];
  float* out = (float*)d_out;

  float* ws = (float*)d_ws;
  size_t off = 0;
  float* xz  = ws + off; off += (size_t)MR * XZS;   // (B,L,1024) u|z
  float* h   = ws + off; off += (size_t)MR * DD;    // residual stream
  float* h2  = ws + off; off += (size_t)MR * DD;    // pre-LN scratch
  float* u2  = ws + off; off += (size_t)MR * DI;    // silu(conv(u))
  float* dbc = ws + off; off += (size_t)MR * DBCS;  // dt_in|Bm|Cm
  float* dtb = ws + off; off += (size_t)MR * DI;    // softplus(dt)

  constexpr int MT = MR / 16;  // 2048 m-tiles

  // 1) input projection (+bias) then LayerNorm(g0,b0)
  //    <NT,K,LDA,N,LDC,SHA,SOFTPLUS>
  wmma_gemm_kernel<2, DIN, DIN, DD, DD, true, false>
      <<<MT, 256, 0, stream>>>(x, Wp, bp, h2);
  layernorm_kernel<<<MR, 256, 0, stream>>>(h2, g0, b0, h);

  for (int l = 0; l < NLAY; ++l) {
    const float* Wi_l  = Wi  + (size_t)l * (2 * DI) * DD;
    const float* cw_l  = cw  + (size_t)l * DI * DC;
    const float* cb_l  = cb  + (size_t)l * DI;
    const float* Wx_l  = Wx  + (size_t)l * DBCS * DI;
    const float* Wdt_l = Wdt + (size_t)l * DI * DTR;
    const float* bdt_l = bdt + (size_t)l * DI;
    const float* Al_l  = Alog+ (size_t)l * DI * DS;
    const float* Dp_l  = Dp  + (size_t)l * DI;
    const float* Wo_l  = Wo  + (size_t)l * DD * DI;
    const float* lg_l  = lng + (size_t)l * DD;
    const float* lb_l  = lnb + (size_t)l * DD;

    // 2) xz = h @ Wi^T            (N=1024, K=256), shared-A, NT=8
    wmma_gemm_kernel<8, DD, DD, XZS, XZS, true, false>
        <<<MT, 256, 0, stream>>>(h, Wi_l, nullptr, xz);

    // 3) u2 = silu(causal depthwise conv(u) + cb)
    conv_silu_kernel<<<16384, 256, 0, stream>>>(xz, cw_l, cb_l, u2, MR * DI);

    // 4) dbc = u2 @ Wx^T          (N=48, K=512), per-wave, NT=3
    wmma_gemm_kernel<3, DI, DI, DBCS, DBCS, false, false>
        <<<MT / 8, 256, 0, stream>>>(u2, Wx_l, nullptr, dbc);

    // 5) dtb = softplus(dt_in @ Wdt^T + bdt)   (N=512, K=16), shared-A
    wmma_gemm_kernel<4, DTR, DBCS, DI, DI, true, true>
        <<<MT, 256, 0, stream>>>(dbc, Wdt_l, bdt_l, dtb);

    // 6) selective scan (fused D-term + silu(z) gate) -> z-half of xz
    scan_kernel<<<BB, DI, 0, stream>>>(u2, dtb, dbc, xz, Al_l, Dp_l);

    // 7) h2 = gated_y @ Wo^T      (N=256, K=512), shared-A, NT=2; then LN
    wmma_gemm_kernel<2, DI, XZS, DD, DD, true, false>
        <<<MT, 256, 0, stream>>>(xz + DI, Wo_l, nullptr, h2);
    layernorm_kernel<<<MR, 256, 0, stream>>>(h2, lg_l, lb_l, h);
  }

  // 8) attention pooling + linear head -> out[B]
  pool_head_kernel<<<BB, 256, 0, stream>>>(h, wa, ba, Wf, bf, out);
}
